// MoE_40398462386786
// MI455X (gfx1250) — matrix operand order repaired and checked by reading, not compiled, not run on hardware
//
#include <hip/hip_runtime.h>
#include <math.h>

// ---------------------------------------------------------------------------
// MoE top-2 FFN for MI455X (gfx1250), wave32 + v_wmma_f32_16x16x32_bf16.
// ---------------------------------------------------------------------------

typedef __attribute__((ext_vector_type(16))) __bf16 v16bf;
typedef __attribute__((ext_vector_type(8)))  __bf16 v8bf;
typedef __attribute__((ext_vector_type(4)))  __bf16 v4bf;
typedef __attribute__((ext_vector_type(8)))  float  v8f;

#define D_MODEL 512
#define D_FF    2048
#define NEXP    8
#define NTOK    8192          // 4 * 2048 tokens
#define XP      520           // padded LDS stride (bf16) for X tile (512+8)
#define HP      136           // padded LDS stride (bf16) for H tile (128+8)

// ----- helpers --------------------------------------------------------------

static __device__ __forceinline__ __bf16 f2bf(float f){
  // round-to-nearest-even f32 -> bf16 (inputs are finite)
  unsigned u = __float_as_uint(f);
  unsigned r = (u + 0x7FFFu + ((u >> 16) & 1u)) >> 16;
  unsigned short s = (unsigned short)r;
  __bf16 b; __builtin_memcpy(&b, &s, 2); return b;
}

static __device__ __forceinline__ float gelu_exact(float v){
  return 0.5f * v * (1.0f + erff(v * 0.70710678118654752440f));
}

// Load 16 bf16 as two 16B segments `gap` elements apart.
// A operand: gap=8  (16 contiguous K values of one row)
// B operand: gap=8  (16 contiguous K values of one column)
static __device__ __forceinline__ v16bf ld16(const __bf16* p, int gap){
  v8bf lo = *(const v8bf*)p;
  v8bf hi = *(const v8bf*)(p + gap);
  return __builtin_shufflevector(lo, hi, 0,1,2,3,4,5,6,7,8,9,10,11,12,13,14,15);
}

static __device__ __forceinline__ v8f wmma_bf16(v16bf a, v16bf b, v8f c){
  return __builtin_amdgcn_wmma_f32_16x16x32_bf16(
      /*neg_a=*/false, a, /*neg_b=*/false, b,
      /*c_mod=*/(short)0, c, /*reuse_a=*/false, /*reuse_b=*/false);
}

// ----- kernel 1: out = x (residual base), zero routing counters -------------

__global__ void init_out_kernel(const float* __restrict__ x,
                                float* __restrict__ out,
                                int* __restrict__ counts){
  size_t i = (size_t)blockIdx.x * blockDim.x + threadIdx.x;   // float4 index
  ((float4*)out)[i] = ((const float4*)x)[i];
  if (blockIdx.x == 0 && threadIdx.x < NEXP) counts[threadIdx.x] = 0;
}

// ----- kernel 2: gating (one wave32 per token) ------------------------------

__global__ void gate_kernel(const float* __restrict__ x,
                            const float* __restrict__ Wg,
                            const float* __restrict__ bg,
                            int* __restrict__ counts,
                            int* __restrict__ tokList,
                            float* __restrict__ gateW){
  const int wid  = threadIdx.x >> 5;
  const int lane = threadIdx.x & 31;
  const int t    = blockIdx.x * 8 + wid;
  const float* xr = x + (size_t)t * D_MODEL;

  float acc[NEXP];
  #pragma unroll
  for (int e = 0; e < NEXP; ++e) acc[e] = 0.f;

  for (int d = lane; d < D_MODEL; d += 32){
    float xv = xr[d];
    float4 w0 = ((const float4*)Wg)[d*2+0];
    float4 w1 = ((const float4*)Wg)[d*2+1];
    acc[0] += xv*w0.x; acc[1] += xv*w0.y; acc[2] += xv*w0.z; acc[3] += xv*w0.w;
    acc[4] += xv*w1.x; acc[5] += xv*w1.y; acc[6] += xv*w1.z; acc[7] += xv*w1.w;
  }
  #pragma unroll
  for (int e = 0; e < NEXP; ++e){
    #pragma unroll
    for (int off = 16; off > 0; off >>= 1)
      acc[e] += __shfl_xor(acc[e], off, 32);
  }
  if (lane == 0){
    float l[NEXP];
    #pragma unroll
    for (int e = 0; e < NEXP; ++e) l[e] = acc[e] + bg[e];
    int i0 = 0;
    #pragma unroll
    for (int e = 1; e < NEXP; ++e) if (l[e] > l[i0]) i0 = e;
    int i1 = (i0 == 0) ? 1 : 0;
    #pragma unroll
    for (int e = 0; e < NEXP; ++e) if (e != i0 && l[e] > l[i1]) i1 = e;
    float dl = expf(l[i1] - l[i0]);          // l[i1] <= l[i0]
    float s0 = 1.f / (1.f + dl);
    float s1 = dl / (1.f + dl);
    int p0 = atomicAdd(&counts[i0], 1);
    tokList[i0*NTOK + p0] = t; gateW[i0*NTOK + p0] = s0;
    int p1 = atomicAdd(&counts[i1], 1);
    tokList[i1*NTOK + p1] = t; gateW[i1*NTOK + p1] = s1;
  }
}

// ----- kernel 3: transpose + f32->bf16 convert: in[e][R][C] -> out[e][C][R] -

__global__ void transpose_cvt_kernel(const float* __restrict__ in,
                                     __bf16* __restrict__ out,
                                     int R, int C){
  __shared__ float t[32][33];
  const int e  = blockIdx.z;
  const int c0 = blockIdx.x * 32;
  const int r0 = blockIdx.y * 32;
  const float* ine  = in  + (size_t)e * R * C;
  __bf16*      oute = out + (size_t)e * R * C;
  const int tx = threadIdx.x, ty = threadIdx.y;     // 32 x 8
  #pragma unroll
  for (int i = 0; i < 4; ++i)
    t[ty + i*8][tx] = ine[(size_t)(r0 + ty + i*8) * C + c0 + tx];
  __syncthreads();
  #pragma unroll
  for (int i = 0; i < 4; ++i)
    oute[(size_t)(c0 + ty + i*8) * R + r0 + tx] = f2bf(t[tx][ty + i*8]);
}

// ----- kernel 4: grouped fused FFN (32-token tile per expert per block) -----

__global__ __launch_bounds__(256) void ffn_kernel(
    const float* __restrict__ x,  const float* __restrict__ b1,
    const float* __restrict__ b2, const __bf16* __restrict__ W1b,
    const __bf16* __restrict__ W2b, const int* __restrict__ counts,
    const int* __restrict__ tokList, const float* __restrict__ gateW,
    float* __restrict__ out)
{
  __shared__ __bf16 Xs[32 * XP];     // gathered token rows, bf16
  __shared__ __bf16 Hs[32 * HP];     // gelu(h) chunk, bf16
  __shared__ int    sTok[32];
  __shared__ float  sW[32];

  const int e    = blockIdx.y;
  const int tile = blockIdx.x;
  const int cnt  = counts[e];
  if (tile * 32 >= cnt) return;      // uniform per block -> EXEC stays full

  const int tid   = threadIdx.x;
  const int lane  = tid & 31;
  const int wid   = tid >> 5;
  const int mt    = wid & 1;         // which 16-row M tile
  const int grp   = wid >> 1;        // 0..3: 128-wide N group / F pair
  const int nlane = lane & 15;
  const int sel8  = (lane >> 4) << 3;   // unused (0 or 8)
  const int sel16 = (lane >> 4) << 4;   // B-operand K offset (0 or 16)
  const int hi8   = (lane >> 4) << 3;   // C/D row offset (0 or 8)

  // stage routing info (tail rows -> token 0, weight 0: contributes 0)
  if (tid < 32){
    int idx = tile * 32 + tid;
    int tok = 0; float w = 0.f;
    if (idx < cnt){ tok = tokList[e*NTOK + idx]; w = gateW[e*NTOK + idx]; }
    sTok[tid] = tok; sW[tid] = w;
  }
  __syncthreads();

  // stage X rows: 32 x 512 f32 -> bf16 LDS (padded stride)
  #pragma unroll
  for (int j = 0; j < 16; ++j){
    int idx = j * 256 + tid;           // 0..4095
    int row = idx >> 7;
    int cv  = idx & 127;
    float4 xv = ((const float4*)x)[(size_t)sTok[row] * (D_MODEL/4) + cv];
    v4bf b; b[0]=f2bf(xv.x); b[1]=f2bf(xv.y); b[2]=f2bf(xv.z); b[3]=f2bf(xv.w);
    *(v4bf*)&Xs[row * XP + cv * 4] = b;
  }
  __syncthreads();

  const __bf16* w1e = W1b + (size_t)e * D_FF * D_MODEL;   // [F][D]
  const __bf16* w2e = W2b + (size_t)e * D_MODEL * D_FF;   // [D][F]

  const v8f vzero = {};
  v8f cacc[8];
  #pragma unroll
  for (int j = 0; j < 8; ++j) cacc[j] = vzero;

  for (int fc = 0; fc < D_FF; fc += 128){
    // GEMM1: H(32x128) = X(32x512) @ W1[:, fc:fc+128]; each wave: 2 H tiles
    v8f hC0 = vzero, hC1 = vzero;
    #pragma unroll 4
    for (int kt = 0; kt < 16; ++kt){
      int k0 = kt * 32;
      v16bf A = ld16(&Xs[(mt*16 + nlane) * XP + k0 + sel16], 8);
      int fg0 = fc + (grp*2 + 0) * 16 + nlane;
      int fg1 = fc + (grp*2 + 1) * 16 + nlane;
      v16bf B0 = ld16(w1e + (size_t)fg0 * D_MODEL + k0 + sel16, 8);
      hC0 = wmma_bf16(A, B0, hC0);
      v16bf B1 = ld16(w1e + (size_t)fg1 * D_MODEL + k0 + sel16, 8);
      hC1 = wmma_bf16(A, B1, hC1);
    }
    __syncthreads();   // previous GEMM2 finished reading Hs
    {
      int col0 = (grp*2 + 0) * 16 + nlane;
      int col1 = (grp*2 + 1) * 16 + nlane;
      float bb0 = b1[(size_t)e * D_FF + fc + col0];
      float bb1 = b1[(size_t)e * D_FF + fc + col1];
      #pragma unroll
      for (int r = 0; r < 8; ++r){
        int row = mt*16 + r + hi8;
        Hs[row * HP + col0] = f2bf(gelu_exact(hC0[r] + bb0));
        Hs[row * HP + col1] = f2bf(gelu_exact(hC1[r] + bb1));
      }
    }
    __syncthreads();
    // GEMM2: C(32x512) += gelu(H)(32x128) @ W2[fc:fc+128, :]
    #pragma unroll
    for (int kk = 0; kk < 4; ++kk){
      v16bf A = ld16(&Hs[(mt*16 + nlane) * HP + kk*32 + sel16], 8);
      #pragma unroll
      for (int j = 0; j < 8; ++j){
        int dg = grp*128 + j*16 + nlane;
        v16bf B = ld16(w2e + (size_t)dg * D_FF + fc + kk*32 + sel16, 8);
        cacc[j] = wmma_bf16(A, B, cacc[j]);
      }
    }
  }

  // epilogue: out[tok] += gate_w * (C + b2)
  int tokr[8]; float wr[8];
  #pragma unroll
  for (int r = 0; r < 8; ++r){
    int rowl = mt*16 + r + hi8;
    tokr[r] = sTok[rowl]; wr[r] = sW[rowl];
  }
  #pragma unroll
  for (int j = 0; j < 8; ++j){
    int dg = grp*128 + j*16 + nlane;
    float b2v = b2[(size_t)e * D_MODEL + dg];
    #pragma unroll
    for (int r = 0; r < 8; ++r){
      atomicAdd(&out[(size_t)tokr[r] * D_MODEL + dg], (cacc[j][r] + b2v) * wr[r]);
    }
  }
}

// ----- workspace layout -----------------------------------------------------

static const size_t OFF_CNT = 0;
static const size_t OFF_TOK = 1024;
static const size_t OFF_GW  = OFF_TOK + (size_t)NEXP * NTOK * sizeof(int);
static const size_t OFF_W1B = (OFF_GW + (size_t)NEXP * NTOK * sizeof(float) + 255) & ~(size_t)255;
static const size_t OFF_W2B = OFF_W1B + (size_t)NEXP * D_FF * D_MODEL * 2;
// total ~= 34 MB

extern "C" void kernel_launch(void* const* d_in, const int* in_sizes, int n_in,
                              void* d_out, int out_size, void* d_ws, size_t ws_size,
                              hipStream_t stream) {
  (void)in_sizes; (void)n_in; (void)out_size; (void)ws_size;
  const float* x  = (const float*)d_in[0];
  const float* Wg = (const float*)d_in[1];
  const float* bg = (const float*)d_in[2];
  const float* W1 = (const float*)d_in[3];
  const float* b1 = (const float*)d_in[4];
  const float* W2 = (const float*)d_in[5];
  const float* b2 = (const float*)d_in[6];
  float* out = (float*)d_out;
  char*  ws  = (char*)d_ws;

  int*    counts  = (int*)   (ws + OFF_CNT);
  int*    tokList = (int*)   (ws + OFF_TOK);
  float*  gateW   = (float*) (ws + OFF_GW);
  __bf16* W1b     = (__bf16*)(ws + OFF_W1B);
  __bf16* W2b     = (__bf16*)(ws + OFF_W2B);

  // 1) residual base + zero counters
  init_out_kernel<<<(NTOK * D_MODEL / 4) / 256, 256, 0, stream>>>(x, out, counts);
  // 2) gating + routing compaction (wave per token)
  gate_kernel<<<NTOK / 8, 256, 0, stream>>>(x, Wg, bg, counts, tokList, gateW);
  // 3) weight transpose + bf16 convert
  transpose_cvt_kernel<<<dim3(D_FF/32, D_MODEL/32, NEXP), dim3(32, 8), 0, stream>>>(
      W1, W1b, D_MODEL, D_FF);   // [D][F] -> [F][D]
  transpose_cvt_kernel<<<dim3(D_MODEL/32, D_FF/32, NEXP), dim3(32, 8), 0, stream>>>(
      W2, W2b, D_FF, D_MODEL);   // [F][D] -> [D][F]
  // 4) grouped fused FFN over routed tokens
  ffn_kernel<<<dim3(NTOK / 32, NEXP), 256, 0, stream>>>(
      x, b1, b2, W1b, W2b, counts, tokList, gateW, out);
}
